// TriangularSelfAttentionBlock_84765474553902
// MI455X (gfx1250) — matrix operand order repaired and checked by reading, not compiled, not run on hardware
//
#include <hip/hip_runtime.h>
#include <hip/hip_bf16.h>

// ---------------------------------------------------------------------------
// MI455X (gfx1250) implementation of the triangular self-attention block.
// All GEMMs run through v_wmma_f32_16x16x32_f16 (wave32 WMMA); the pair-MLP
// streams its 128MB operand through the Tensor Data Mover (TDM) with
// double-buffered LDS tiles.
// ---------------------------------------------------------------------------

typedef __attribute__((ext_vector_type(8)))  _Float16     v8h;
typedef __attribute__((ext_vector_type(16))) _Float16     v16h;
typedef __attribute__((ext_vector_type(8)))  float        v8f;
typedef __attribute__((ext_vector_type(4)))  unsigned int v4u;
typedef __attribute__((ext_vector_type(4)))  int          v4i;
typedef __attribute__((ext_vector_type(8)))  int          v8i;
typedef _Float16 half_t;

#define WMMA_F16(a, b, c) \
  __builtin_amdgcn_wmma_f32_16x16x32_f16(false, (a), false, (b), (short)0, (c), false, false)

#if defined(__has_builtin)
#if __has_builtin(__builtin_amdgcn_tensor_load_to_lds) && \
    __has_builtin(__builtin_amdgcn_s_wait_tensorcnt)
#define HAVE_TDM 1
#endif
#endif
#ifndef HAVE_TDM
#define HAVE_TDM 0
#endif

// A fragment (16x32 f16, MxK): lane holds row m = lane%16,
// 16 contiguous K values starting at c = (lane<16 ? 0 : 16).
__device__ __forceinline__ v16h load_a_frag(const half_t* A, int lda) {
  const int lane = threadIdx.x & 31;
  const int m = lane & 15;
  const int c = (lane & 16) ? 16 : 0;
  const half_t* p = A + (size_t)m * lda + c;
  return *(const v16h*)(p);
}

// B fragment (32x16 f16, KxN) with B stored row-major as (N,K):
// lane holds column n = lane%16, 16 contiguous K values starting at
// k0 = (lane<16 ? 0 : 16).
__device__ __forceinline__ v16h load_b_frag(const half_t* Bnk, int ldb) {
  const int lane = threadIdx.x & 31;
  const int n = lane & 15;
  const int k0 = (lane & 16) ? 16 : 0;
  return *(const v16h*)(Bnk + (size_t)n * ldb + k0);
}

// ---------------------------------------------------------------------------
// TDM: async load of one contiguous tile (nelem f32) from global to LDS.
// D# layout per CDNA5 ISA §8 (group0: count/lds/global/type, group1: 1-D tile).
// Issued by thread 0 only (TDM ignores EXEC; one issue per wave).
// ---------------------------------------------------------------------------
__device__ __forceinline__ void tile_load_async(float* lds_dst, const float* gsrc, int nelem) {
#if HAVE_TDM
  if (threadIdx.x == 0) {
    const unsigned long long ga = (unsigned long long)(void*)gsrc;  // 57-bit global byte addr
    const unsigned la = (unsigned)(unsigned long long)(void*)lds_dst; // flat low32 == LDS addr
    v4u g0;
    g0[0] = 1u;                                                  // count=1, user desc
    g0[1] = la;                                                  // lds_addr [63:32]
    g0[2] = (unsigned)ga;                                        // global_addr [95:64]
    g0[3] = ((unsigned)(ga >> 32) & 0x01FFFFFFu) | 0x80000000u;  // addr[56:32] | type=2
    v8i g1;
    g1[0] = 0x00020000;                       // data_size=4B, no multicast/pad/iterate
    g1[1] = (nelem & 0xFFFF) << 16;           // tensor_dim0[15:0]  (bits 63:48)
    g1[2] = ((nelem >> 16) & 0xFFFF) | (1 << 16); // tensor_dim0[31:16] | tensor_dim1=1
    g1[3] = (nelem & 0xFFFF) << 16;           // tile_dim0 (bits 127:112)
    g1[4] = 0;                                // tile_dim1=0, tile_dim2=0
    g1[5] = nelem;                            // tensor_dim0_stride (low 32)
    g1[6] = 0;
    g1[7] = 0;
    v4i gz = {0, 0, 0, 0};
#if __clang_major__ >= 23
    v8i gz8 = {0, 0, 0, 0, 0, 0, 0, 0};
    __builtin_amdgcn_tensor_load_to_lds(g0, g1, gz, gz, gz8, 0);
#else
    __builtin_amdgcn_tensor_load_to_lds(g0, g1, gz, gz, 0);
#endif
  }
#else
  for (int i = threadIdx.x; i < nelem; i += blockDim.x) lds_dst[i] = gsrc[i];
#endif
}

// Wait until at most `n` TDM ops outstanding (wave 0 only; barrier publishes).
__device__ __forceinline__ void tile_wait(int n) {
#if HAVE_TDM
  if (threadIdx.x == 0) {
    if (n == 0) __builtin_amdgcn_s_wait_tensorcnt((short)0);
    else        __builtin_amdgcn_s_wait_tensorcnt((short)1);
  }
#else
  (void)n;
#endif
}

// ---------------------------------------------------------------------------
// fp32 -> fp16 conversion (weights)
// ---------------------------------------------------------------------------
__global__ void k_cvt_f32_f16(const float* __restrict__ x, half_t* __restrict__ y, long long n) {
  long long i = (long long)blockIdx.x * blockDim.x + threadIdx.x;
  if (i < n) y[i] = (half_t)x[i];
}

// ---------------------------------------------------------------------------
// LayerNorm over last dim D, one wave per row, f16 output.
// ---------------------------------------------------------------------------
__global__ void __launch_bounds__(256) k_ln_rows(
    const float* __restrict__ x, const float* __restrict__ g, const float* __restrict__ b,
    half_t* __restrict__ out, int rows, int D)
{
  const int row = blockIdx.x * (blockDim.x >> 5) + (threadIdx.x >> 5);
  if (row >= rows) return;
  const int lane = threadIdx.x & 31;
  const float* xr = x + (size_t)row * D;
  float s = 0.f, s2 = 0.f;
  for (int i = lane; i < D; i += 32) { float v = xr[i]; s += v; s2 += v * v; }
#pragma unroll
  for (int off = 16; off; off >>= 1) {
    s  += __shfl_xor(s,  off, 32);
    s2 += __shfl_xor(s2, off, 32);
  }
  const float mean = s / D;
  const float inv  = rsqrtf(fmaxf(s2 / D - mean * mean, 0.f) + 1e-5f);
  half_t* orow = out + (size_t)row * D;
  for (int i = lane; i < D; i += 32)
    orow[i] = (half_t)((xr[i] - mean) * inv * g[i] + b[i]);
}

// ---------------------------------------------------------------------------
// Fused pairwise LN + bias GEMM:  bias[i*512+j, h] = LN(pair[i,j,:]) @ W_p2s.T
// One wave handles 16 flattened (i,j) rows. K=128, N=32.
// ---------------------------------------------------------------------------
__global__ void __launch_bounds__(32) k_pair_bias(
    const float* __restrict__ pair, const float* __restrict__ g, const float* __restrict__ bvec,
    const half_t* __restrict__ Wh /*32x128*/, float* __restrict__ biasb /*[M x 32]*/)
{
  __shared__ half_t xh[16 * 128];
  const size_t r0 = (size_t)blockIdx.x * 16;
  const int lane = threadIdx.x;
  const int row = lane >> 1, hlf = lane & 1;            // 2 lanes per row
  const float* pr = pair + (r0 + row) * 128;
  float s = 0.f, s2 = 0.f;
  for (int c = hlf * 64; c < hlf * 64 + 64; ++c) { float v = pr[c]; s += v; s2 += v * v; }
  s  += __shfl_xor(s,  1, 32);
  s2 += __shfl_xor(s2, 1, 32);
  const float mean = s * (1.f / 128.f);
  const float inv  = rsqrtf(fmaxf(s2 * (1.f / 128.f) - mean * mean, 0.f) + 1e-5f);
  for (int c = hlf * 64; c < hlf * 64 + 64; ++c)
    xh[row * 128 + c] = (half_t)((pr[c] - mean) * inv * g[c] + bvec[c]);
  __syncthreads();

  const int nn = lane & 15, mo = (lane & 16) ? 8 : 0;
#pragma unroll
  for (int nt = 0; nt < 2; ++nt) {
    v8f acc = {};
#pragma unroll
    for (int k = 0; k < 128; k += 32) {
      v16h a = load_a_frag(xh + k, 128);
      v16h b = load_b_frag(Wh + (size_t)(nt * 16) * 128 + k, 128);
      acc = WMMA_F16(a, b, acc);
    }
#pragma unroll
    for (int v = 0; v < 8; ++v)
      biasb[(r0 + mo + v) * 32 + nt * 16 + nn] = acc[v];
  }
}

// ---------------------------------------------------------------------------
// Generic WMMA GEMM: C[M,N] = act(A[M,K] @ B[N,K]^T + bias) + residual
// Block = 4 waves stacked on M (64 rows); each wave owns 16 x (16*NT) cols.
// act: 0=none, 1=relu, 2=sigmoid.
// ---------------------------------------------------------------------------
template <int NT>
__global__ void __launch_bounds__(128) k_gemm(
    const half_t* __restrict__ A, int lda,
    const half_t* __restrict__ B, int ldb,
    int M, int N, int K,
    const float* __restrict__ bias,
    const float* __restrict__ residual, int ldr,
    float* __restrict__ Cf, half_t* __restrict__ Ch, int ldc, int act)
{
  (void)M; (void)N;
  const int wave = threadIdx.x >> 5;
  const int m0 = blockIdx.y * 64 + wave * 16;
  const int n0 = blockIdx.x * (16 * NT);
  v8f acc[NT] = {};
  const half_t* Abase = A + (size_t)m0 * lda;
  for (int k = 0; k < K; k += 32) {
    if (k + 64 < K) __builtin_prefetch(Abase + k + 64, 0, 1);   // global_prefetch_b8
    v16h av = load_a_frag(Abase + k, lda);
#pragma unroll
    for (int t = 0; t < NT; ++t) {
      v16h bv = load_b_frag(B + (size_t)(n0 + t * 16) * ldb + k, ldb);
      acc[t] = WMMA_F16(av, bv, acc[t]);
    }
  }
  const int lane = threadIdx.x & 31;
  const int nn = lane & 15, mo = (lane & 16) ? 8 : 0;
#pragma unroll
  for (int t = 0; t < NT; ++t) {
    const int col = n0 + t * 16 + nn;
#pragma unroll
    for (int v = 0; v < 8; ++v) {
      const int row = m0 + mo + v;
      float val = acc[t][v];
      if (bias) val += bias[col];
      if (act == 1)      val = fmaxf(val, 0.f);
      else if (act == 2) val = 1.f / (1.f + __expf(-val));
      if (residual) val += residual[(size_t)row * ldr + col];
      if (Cf) Cf[(size_t)row * ldc + col] = val;
      if (Ch) Ch[(size_t)row * ldc + col] = (half_t)val;
    }
  }
}

// ---------------------------------------------------------------------------
// Transpose V slices: vT[h][c][j] = qkv[j][h*96 + 64 + c]
// ---------------------------------------------------------------------------
__global__ void k_vT(const half_t* __restrict__ qkv, half_t* __restrict__ vT) {
  int idx = blockIdx.x * blockDim.x + threadIdx.x;
  if (idx >= 32 * 32 * 512) return;
  const int j = idx & 511, c = (idx >> 9) & 31, h = idx >> 14;
  vT[idx] = qkv[(size_t)j * 3072 + h * 96 + 64 + c];
}

// ---------------------------------------------------------------------------
// Attention scores + softmax: one wave per (head, 16-row i-tile).
// ---------------------------------------------------------------------------
__global__ void __launch_bounds__(32) k_attn_scores(
    const half_t* __restrict__ qkv, const float* __restrict__ biasb,
    const unsigned char* __restrict__ mask, half_t* __restrict__ probs)
{
  __shared__ float srow[16 * 512];
  const int h = blockIdx.y, i0 = blockIdx.x * 16;
  const int lane = threadIdx.x;
  const float scale = 0.17677669529663687f;  // 1/sqrt(32)
  v16h qv = load_a_frag(qkv + (size_t)i0 * 3072 + h * 96, 3072);  // K=32 in one frag
  const int nn = lane & 15, mo = (lane & 16) ? 8 : 0;
  for (int jt = 0; jt < 32; ++jt) {
    v16h kv = load_b_frag(qkv + (size_t)(jt * 16) * 3072 + h * 96 + 32, 3072);
    v8f acc = {};
    acc = WMMA_F16(qv, kv, acc);
#pragma unroll
    for (int v = 0; v < 8; ++v) {
      const int r = mo + v, col = jt * 16 + nn;
      float val = acc[v] * scale + biasb[((size_t)(i0 + r) * 512 + col) * 32 + h];
      if (!mask[col]) val = -1e30f;
      srow[r * 512 + col] = val;
    }
  }
  __syncthreads();
  const int r = lane & 15, hlf = lane >> 4;
  const float* sr = srow + r * 512 + hlf * 256;
  float m = -1e30f;
  for (int c = 0; c < 256; ++c) m = fmaxf(m, sr[c]);
  m = fmaxf(m, __shfl_xor(m, 16, 32));
  float sum = 0.f;
  for (int c = 0; c < 256; ++c) sum += __expf(sr[c] - m);
  sum += __shfl_xor(sum, 16, 32);
  const float rinv = 1.f / sum;
  half_t* prow = probs + ((size_t)h * 512 + i0 + r) * 512 + hlf * 256;
  for (int c = 0; c < 256; ++c) prow[c] = (half_t)(__expf(sr[c] - m) * rinv);
}

// ---------------------------------------------------------------------------
// o[i, h*32+c] = sum_j probs[h,i,j] * v[j,c]; one wave per (head, i-tile).
// ---------------------------------------------------------------------------
__global__ void __launch_bounds__(32) k_attn_out(
    const half_t* __restrict__ probs, const half_t* __restrict__ vT, float* __restrict__ o)
{
  const int h = blockIdx.y, i0 = blockIdx.x * 16;
  const half_t* A = probs + ((size_t)h * 512 + i0) * 512;
  const half_t* B = vT + (size_t)h * 32 * 512;
  v8f acc0 = {}, acc1 = {};
  for (int k = 0; k < 512; k += 32) {
    v16h a  = load_a_frag(A + k, 512);
    v16h b0 = load_b_frag(B + k, 512);
    v16h b1 = load_b_frag(B + (size_t)16 * 512 + k, 512);
    acc0 = WMMA_F16(a, b0, acc0);
    acc1 = WMMA_F16(a, b1, acc1);
  }
  const int lane = threadIdx.x & 31, nn = lane & 15, mo = (lane & 16) ? 8 : 0;
#pragma unroll
  for (int v = 0; v < 8; ++v) {
    const int row = i0 + mo + v;
    o[(size_t)row * 1024 + h * 32 + nn]      = acc0[v];
    o[(size_t)row * 1024 + h * 32 + 16 + nn] = acc1[v];
  }
}

// ---------------------------------------------------------------------------
// Gated combine: og = sigmoid(y Wg + bg) * attn_out  (f16 out)
// ---------------------------------------------------------------------------
__global__ void k_gate_mul(const float* __restrict__ g, const float* __restrict__ o,
                           half_t* __restrict__ out, int n) {
  int i = blockIdx.x * blockDim.x + threadIdx.x;
  if (i < n) out[i] = (half_t)(g[i] * o[i]);
}

// ---------------------------------------------------------------------------
// SequenceToPair: p[i,j,:] = pair[i,j,:] + [qh[j]*kh[i], qh[j]-kh[i]] @ Wspo^T + bspo
// ---------------------------------------------------------------------------
__global__ void __launch_bounds__(32) k_seq2pair(
    const half_t* __restrict__ sp, const float* __restrict__ pair,
    const half_t* __restrict__ Wspo, const float* __restrict__ bspo, float* __restrict__ p)
{
  __shared__ half_t xh[16 * 128];
  const int i = blockIdx.y, j0 = blockIdx.x * 16;
  const int lane = threadIdx.x;
  for (int t = lane; t < 16 * 128; t += 32) {
    const int r = t >> 7, c = t & 127, c64 = c & 63;
    const float qv = (float)sp[(size_t)(j0 + r) * 128 + c64];
    const float kv = (float)sp[(size_t)i * 128 + 64 + c64];
    xh[t] = (half_t)((c < 64) ? qv * kv : qv - kv);
  }
  __syncthreads();
  const int nn = lane & 15, mo = (lane & 16) ? 8 : 0;
  for (int nt = 0; nt < 8; ++nt) {
    v8f acc = {};
#pragma unroll
    for (int k = 0; k < 128; k += 32) {
      v16h a = load_a_frag(xh + k, 128);
      v16h b = load_b_frag(Wspo + (size_t)(nt * 16) * 128 + k, 128);
      acc = WMMA_F16(a, b, acc);
    }
#pragma unroll
    for (int v = 0; v < 8; ++v) {
      const int j = j0 + mo + v, col = nt * 16 + nn;
      const size_t idx = ((size_t)i * 512 + j) * 128 + col;
      p[idx] = pair[idx] + acc[v] + bspo[col];
    }
  }
}

// ---------------------------------------------------------------------------
// Fused pair MLP: p += relu(LN(p) @ Wp1^T + bp1) @ Wp2^T + bp2
// Block = 128 threads (4 waves). Each workgroup owns TPW consecutive 16-row
// tiles; the p tile for step t+1 is TDM-loaded into LDS while step t computes
// (double buffer, s_wait_tensorcnt pipelining). Residual comes from LDS.
// ---------------------------------------------------------------------------
static constexpr int TPW = 8;  // tiles per workgroup

__global__ void __launch_bounds__(128) k_mlp_pair(
    float* __restrict__ p, const float* __restrict__ lng, const float* __restrict__ lnb,
    const half_t* __restrict__ Wp1, const float* __restrict__ bp1,
    const half_t* __restrict__ Wp2, const float* __restrict__ bp2)
{
  __shared__ float  ptile[2][16 * 128];
  __shared__ half_t xh[16 * 128];
  __shared__ half_t hh[16 * 512];
  const int tid = threadIdx.x;
  const size_t base = (size_t)blockIdx.x * TPW * 16;   // first row of first tile

  tile_load_async(ptile[0], p + base * 128, 16 * 128);

  for (int t = 0; t < TPW; ++t) {
    const size_t r0 = base + (size_t)t * 16;
    if (t + 1 < TPW)
      tile_load_async(ptile[(t + 1) & 1], p + (r0 + 16) * 128, 16 * 128);
    tile_wait((t + 1 < TPW) ? 1 : 0);   // tile t resident (TDM in-order per wave)
    __syncthreads();

    const float* pt = ptile[t & 1];
    // --- LayerNorm (8 threads per row) from LDS ---
    {
      const int row = tid >> 3, sub = tid & 7;
      const float* pr = pt + row * 128;
      float s = 0.f, s2 = 0.f;
      for (int c = sub; c < 128; c += 8) { float v = pr[c]; s += v; s2 += v * v; }
      s += __shfl_xor(s, 4, 32);  s2 += __shfl_xor(s2, 4, 32);
      s += __shfl_xor(s, 2, 32);  s2 += __shfl_xor(s2, 2, 32);
      s += __shfl_xor(s, 1, 32);  s2 += __shfl_xor(s2, 1, 32);
      const float mean = s * (1.f / 128.f);
      const float inv  = rsqrtf(fmaxf(s2 * (1.f / 128.f) - mean * mean, 0.f) + 1e-5f);
      for (int c = sub; c < 128; c += 8)
        xh[row * 128 + c] = (half_t)((pr[c] - mean) * inv * lng[c] + lnb[c]);
    }
    __syncthreads();

    const int wave = tid >> 5, lane = tid & 31;
    const int nn = lane & 15, mo = (lane & 16) ? 8 : 0;
    // --- GEMM1: 16x128 @ 128x512, relu -> LDS (each wave: 8 column tiles) ---
    for (int tt = 0; tt < 8; ++tt) {
      const int nt = wave * 8 + tt;
      v8f acc = {};
#pragma unroll
      for (int k = 0; k < 128; k += 32) {
        v16h a = load_a_frag(xh + k, 128);
        v16h b = load_b_frag(Wp1 + (size_t)(nt * 16) * 128 + k, 128);
        acc = WMMA_F16(a, b, acc);
      }
#pragma unroll
      for (int v = 0; v < 8; ++v) {
        const int col = nt * 16 + nn;
        hh[(mo + v) * 512 + col] = (half_t)fmaxf(acc[v] + bp1[col], 0.f);
      }
    }
    __syncthreads();
    // --- GEMM2: 16x512 @ 512x128, residual (from LDS) -> global p ---
    for (int tt = 0; tt < 2; ++tt) {
      const int nt = wave * 2 + tt;
      v8f acc = {};
      for (int k = 0; k < 512; k += 32) {
        v16h a = load_a_frag(hh + k, 512);
        v16h b = load_b_frag(Wp2 + (size_t)(nt * 16) * 512 + k, 512);
        acc = WMMA_F16(a, b, acc);
      }
#pragma unroll
      for (int v = 0; v < 8; ++v) {
        const int col = nt * 16 + nn;
        p[(r0 + mo + v) * 128 + col] = pt[(mo + v) * 128 + col] + acc[v] + bp2[col];
      }
    }
    __syncthreads();   // all reads of ptile[t&1] done before it is refilled
  }
}

// ---------------------------------------------------------------------------
// Host orchestration
// ---------------------------------------------------------------------------
extern "C" void kernel_launch(void* const* d_in, const int* in_sizes, int n_in,
                              void* d_out, int out_size, void* d_ws, size_t ws_size,
                              hipStream_t stream)
{
  (void)in_sizes; (void)n_in; (void)out_size; (void)ws_size;
  const float* seq    = (const float*)d_in[0];
  const float* pair   = (const float*)d_in[1];
  const float* lnp2sg = (const float*)d_in[2];
  const float* lnp2sb = (const float*)d_in[3];
  const float* W_p2s  = (const float*)d_in[4];
  const float* ln1g   = (const float*)d_in[5];
  const float* ln1b   = (const float*)d_in[6];
  const float* Wqkv   = (const float*)d_in[7];
  const float* Wo     = (const float*)d_in[8];
  const float* bo     = (const float*)d_in[9];
  const float* Wg     = (const float*)d_in[10];
  const float* bg     = (const float*)d_in[11];
  const float* lnsg   = (const float*)d_in[12];
  const float* lnsb   = (const float*)d_in[13];
  const float* Wm1    = (const float*)d_in[14];
  const float* bm1    = (const float*)d_in[15];
  const float* Wm2    = (const float*)d_in[16];
  const float* bm2    = (const float*)d_in[17];
  const float* lnspg  = (const float*)d_in[18];
  const float* lnspb  = (const float*)d_in[19];
  const float* Wsp    = (const float*)d_in[20];
  const float* bsp    = (const float*)d_in[21];
  const float* Wspo   = (const float*)d_in[22];
  const float* bspo   = (const float*)d_in[23];
  const float* lnmpg  = (const float*)d_in[24];
  const float* lnmpb  = (const float*)d_in[25];
  const float* Wp1    = (const float*)d_in[26];
  const float* bp1    = (const float*)d_in[27];
  const float* Wp2    = (const float*)d_in[28];
  const float* bp2    = (const float*)d_in[29];
  const unsigned char* mask = (const unsigned char*)d_in[30];

  float* s_out = (float*)d_out;
  float* p_out = s_out + (size_t)512 * 1024;

  char* wsb = (char*)d_ws;
  size_t off = 0;
  auto alloc = [&](size_t bytes) -> void* {
    void* p = wsb + off;
    off += (bytes + 255) & ~(size_t)255;
    return p;
  };

  half_t* Wp2s_h = (half_t*)alloc((size_t)32 * 128 * 2);
  half_t* Wqkv_h = (half_t*)alloc((size_t)3072 * 1024 * 2);
  half_t* Wo_h   = (half_t*)alloc((size_t)1024 * 1024 * 2);
  half_t* Wg_h   = (half_t*)alloc((size_t)1024 * 1024 * 2);
  half_t* Wm1_h  = (half_t*)alloc((size_t)4096 * 1024 * 2);
  half_t* Wm2_h  = (half_t*)alloc((size_t)1024 * 4096 * 2);
  half_t* Wsp_h  = (half_t*)alloc((size_t)128 * 1024 * 2);
  half_t* Wspo_h = (half_t*)alloc((size_t)128 * 128 * 2);
  half_t* Wp1_h  = (half_t*)alloc((size_t)512 * 128 * 2);
  half_t* Wp2_h  = (half_t*)alloc((size_t)128 * 512 * 2);
  float*  biasb  = (float*)alloc((size_t)512 * 512 * 32 * 4);   // [i*512+j, h]
  half_t* y_h    = (half_t*)alloc((size_t)512 * 1024 * 2);
  half_t* qkv_h  = (half_t*)alloc((size_t)512 * 3072 * 2);
  half_t* vT     = (half_t*)alloc((size_t)32 * 32 * 512 * 2);
  half_t* probs  = (half_t*)alloc((size_t)32 * 512 * 512 * 2);
  float*  gbuf   = (float*)alloc((size_t)512 * 1024 * 4);
  float*  obuf   = (float*)alloc((size_t)512 * 1024 * 4);
  half_t* og_h   = (half_t*)alloc((size_t)512 * 1024 * 2);
  float*  s1     = (float*)alloc((size_t)512 * 1024 * 4);
  half_t* snh    = (half_t*)alloc((size_t)512 * 1024 * 2);
  half_t* h1     = (half_t*)alloc((size_t)512 * 4096 * 2);
  half_t* sp_h   = (half_t*)alloc((size_t)512 * 128 * 2);

  auto cvt = [&](const float* s, half_t* d, long long n) {
    k_cvt_f32_f16<<<dim3((unsigned)((n + 255) / 256)), dim3(256), 0, stream>>>(s, d, n);
  };
  cvt(W_p2s, Wp2s_h, 32LL * 128);
  cvt(Wqkv,  Wqkv_h, 3072LL * 1024);
  cvt(Wo,    Wo_h,   1024LL * 1024);
  cvt(Wg,    Wg_h,   1024LL * 1024);
  cvt(Wm1,   Wm1_h,  4096LL * 1024);
  cvt(Wm2,   Wm2_h,  1024LL * 4096);
  cvt(Wsp,   Wsp_h,  128LL * 1024);
  cvt(Wspo,  Wspo_h, 128LL * 128);
  cvt(Wp1,   Wp1_h,  512LL * 128);
  cvt(Wp2,   Wp2_h,  128LL * 512);

  // PairToSequence: fused LN + bias GEMM over 262144 rows.
  k_pair_bias<<<dim3(16384), dim3(32), 0, stream>>>(pair, lnp2sg, lnp2sb, Wp2s_h, biasb);

  // y = LN(seq)
  k_ln_rows<<<dim3(64), dim3(256), 0, stream>>>(seq, ln1g, ln1b, y_h, 512, 1024);

  // qkv = y @ Wqkv^T  (f16 out)
  k_gemm<4><<<dim3(48, 8), dim3(128), 0, stream>>>(y_h, 1024, Wqkv_h, 1024, 512, 3072, 1024,
      nullptr, nullptr, 0, nullptr, qkv_h, 3072, 0);

  k_vT<<<dim3(2048), dim3(256), 0, stream>>>(qkv_h, vT);
  k_attn_scores<<<dim3(32, 32), dim3(32), 0, stream>>>(qkv_h, biasb, mask, probs);
  k_attn_out<<<dim3(32, 32), dim3(32), 0, stream>>>(probs, vT, obuf);

  // gate = sigmoid(y @ Wg^T + bg)
  k_gemm<4><<<dim3(16, 8), dim3(128), 0, stream>>>(y_h, 1024, Wg_h, 1024, 512, 1024, 1024,
      bg, nullptr, 0, gbuf, nullptr, 1024, 2);
  k_gate_mul<<<dim3(2048), dim3(256), 0, stream>>>(gbuf, obuf, og_h, 512 * 1024);

  // s1 = seq + (gate*o) @ Wo^T + bo
  k_gemm<4><<<dim3(16, 8), dim3(128), 0, stream>>>(og_h, 1024, Wo_h, 1024, 512, 1024, 1024,
      bo, seq, 1024, s1, nullptr, 1024, 0);

  // mlp_seq: s = s1 + relu(LN(s1) Wm1 + bm1) Wm2 + bm2   -> d_out[s]
  k_ln_rows<<<dim3(64), dim3(256), 0, stream>>>(s1, lnsg, lnsb, snh, 512, 1024);
  k_gemm<4><<<dim3(64, 8), dim3(128), 0, stream>>>(snh, 1024, Wm1_h, 1024, 512, 4096, 1024,
      bm1, nullptr, 0, nullptr, h1, 4096, 1);
  k_gemm<4><<<dim3(16, 8), dim3(128), 0, stream>>>(h1, 4096, Wm2_h, 4096, 512, 1024, 4096,
      bm2, s1, 1024, s_out, nullptr, 1024, 0);

  // SequenceToPair projections
  k_ln_rows<<<dim3(64), dim3(256), 0, stream>>>(s_out, lnspg, lnspb, snh, 512, 1024);
  k_gemm<4><<<dim3(2, 8), dim3(128), 0, stream>>>(snh, 1024, Wsp_h, 1024, 512, 128, 1024,
      bsp, nullptr, 0, nullptr, sp_h, 128, 0);
  k_seq2pair<<<dim3(32, 512), dim3(32), 0, stream>>>(sp_h, pair, Wspo_h, bspo, p_out);

  // mlp_pair (fused, TDM double-buffered, in-place on d_out[p])
  k_mlp_pair<<<dim3(16384 / TPW), dim3(128), 0, stream>>>(
      p_out, lnmpg, lnmpb, Wp1_h, bp1, Wp2_h, bp2);
}
